// ContextualAttention_66365834658271
// MI455X (gfx1250) — matrix operand order repaired and checked by reading, not compiled, not run on hardware
//
#include <hip/hip_runtime.h>
#include <hip/hip_bf16.h>
#include <cstdint>

typedef __attribute__((ext_vector_type(16))) _Float16 v16h;
typedef __attribute__((ext_vector_type(8)))  _Float16 v8h;
typedef __attribute__((ext_vector_type(8)))  float    v8f;
typedef __attribute__((ext_vector_type(4)))  int      v4i;

#define BN   4
#define C_   64
#define H_   96
#define W_   96
#define D_   48
#define L_   2304      // 48*48
#define K1_  576       // 64*9
#define HW_  9216      // 96*96
#define LDK  40        // LDS row stride in halves (32 + 8 pad)

#if defined(__gfx1250__) && __has_builtin(__builtin_amdgcn_global_load_async_to_lds_b128) && __has_builtin(__builtin_amdgcn_s_wait_asynccnt)
#define ASYNC_CP 1
#else
#define ASYNC_CP 0
#endif

#if ASYNC_CP
typedef __attribute__((address_space(1))) v4i GV4;
typedef __attribute__((address_space(3))) v4i LV4;
#endif

__device__ __forceinline__ int swp(int i) { return (i % 48) * 48 + (i / 48); }

__device__ __forceinline__ v16h cat8(v8h lo, v8h hi) {
  return __builtin_shufflevector(lo, hi, 0,1,2,3,4,5,6,7,8,9,10,11,12,13,14,15);
}

__device__ __forceinline__ v8f wmma_f16(v16h a, v16h b, v8f c) {
  return __builtin_amdgcn_wmma_f32_16x16x32_f16(false, a, false, b, (short)0, c, false, false);
}

// 16-byte global -> LDS copy; async (ASYNCcnt-tracked, no VGPR round trip) when available.
__device__ __forceinline__ void cp16(_Float16* ldst, const _Float16* gsrc) {
#if ASYNC_CP
  __builtin_amdgcn_global_load_async_to_lds_b128(
      (GV4*)(uintptr_t)gsrc,
      (LV4*)ldst,
      0, 0);
#else
  *(v8h*)ldst = *(const v8h*)gsrc;
#endif
}

__device__ __forceinline__ void cp_commit() {
#if ASYNC_CP
  __builtin_amdgcn_s_wait_asynccnt(0);
#endif
}

// ---------------- gather / conversion kernels ----------------

// Xcol[item][s][k] = fd_pad[c', y+u-1, x+v-1]  (im2row of downsampled f, 3x3 pad 1)
__global__ void k_build_xcol(const float* __restrict__ f, _Float16* __restrict__ X) {
  size_t tid = (size_t)blockIdx.x * blockDim.x + threadIdx.x;
  if (tid >= (size_t)BN * L_ * K1_) return;
  int k    = tid % K1_;
  int s    = (tid / K1_) % L_;
  int item = tid / ((size_t)K1_ * L_);
  int y = s / 48, x = s % 48;
  int c = k / 9, t = k % 9, u = t / 3, v = t % 3;
  int yy = y + u - 1, xx = x + v - 1;
  float val = 0.f;
  if ((unsigned)yy < 48u && (unsigned)xx < 48u)
    val = f[(((size_t)item * C_ + c) * H_ + 2 * yy) * W_ + 2 * xx];
  X[tid] = (_Float16)val;
}

// Wn[item][p][k]: gather patch-p weights (channel p/36, flat slice (p%36)*576+k of
// per-channel (i,j,u,v) flattening of 3x3 patches of downsampled b), L2-normalize row.
__global__ void k_build_wn(const float* __restrict__ b, _Float16* __restrict__ Wn) {
  int lane = threadIdx.x & 31;
  int wv   = threadIdx.x >> 5;
  int pg   = blockIdx.x * 8 + wv;
  int item = pg / L_;
  int p    = pg % L_;
  int c    = p / 36;
  int base = (p % 36) * 576;
  float vals[18];
  float ss = 0.f;
#pragma unroll
  for (int j = 0; j < 18; ++j) {
    int k = lane + 32 * j;
    int flat = base + k;
    int i  = flat / 432;
    int r  = flat % 432;
    int jc = r / 9;
    int t  = r % 9;
    int u = t / 3, v = t % 3;
    int yy = i + u - 1, xx = jc + v - 1;
    float val = 0.f;
    if ((unsigned)yy < 48u && (unsigned)xx < 48u)
      val = b[(((size_t)item * C_ + c) * H_ + 2 * yy) * W_ + 2 * xx];
    vals[j] = val;
    ss += val * val;
  }
#pragma unroll
  for (int off = 16; off > 0; off >>= 1) ss += __shfl_xor(ss, off, 32);
  float nrm = sqrtf(ss);
  float sc  = 1.f / fmaxf(nrm, 1e-4f);
  size_t rowo = ((size_t)item * L_ + p) * K1_;
#pragma unroll
  for (int j = 0; j < 18; ++j)
    Wn[rowo + lane + 32 * j] = (_Float16)(vals[j] * sc);
}

// RWt[item][tap=(u,v)][cp][p] = 0.25 * raw_w[p,cp,3-u,3-v] gathered from b (4x4/stride2 patches)
__global__ void k_build_rwt(const float* __restrict__ b, _Float16* __restrict__ RWt) {
  size_t tid = (size_t)blockIdx.x * blockDim.x + threadIdx.x;
  if (tid >= (size_t)BN * 16 * C_ * L_) return;
  int p    = tid % L_;
  size_t rest = tid / L_;
  int cp   = rest % C_; rest /= C_;
  int uv   = rest % 16;
  int item = rest / 16;
  int u = uv / 4, v = uv % 4;
  int kk = cp * 16 + (3 - u) * 4 + (3 - v);
  int c  = p / 36;
  int flat = (p % 36) * 1024 + kk;
  int i  = flat / 768;
  int r  = flat % 768;
  int j  = r / 16;
  int t2 = r % 16;
  int u2 = t2 / 4, v2 = t2 % 4;
  int yy = 2 * i + u2 - 1, xx = 2 * j + v2 - 1;
  float val = 0.f;
  if ((unsigned)yy < 96u && (unsigned)xx < 96u)
    val = b[(((size_t)item * C_ + c) * H_ + yy) * W_ + xx];
  RWt[tid] = (_Float16)(0.25f * val);
}

__global__ void k_wcvt(const float* __restrict__ w1, const float* __restrict__ w2,
                       _Float16* __restrict__ Wh) {
  int tid = blockIdx.x * blockDim.x + threadIdx.x;
  if (tid >= 2 * C_ * K1_) return;
  float v = (tid < C_ * K1_) ? w1[tid] : w2[tid - C_ * K1_];
  Wh[tid] = (_Float16)v;
}

// im2row 3x3 pad1 of a (BN,64,96,96) f32 tensor -> (BN, 9216, 576) f16
__global__ void k_im2row(const float* __restrict__ src, _Float16* __restrict__ R) {
  size_t tid = (size_t)blockIdx.x * blockDim.x + threadIdx.x;
  if (tid >= (size_t)BN * HW_ * K1_) return;
  int k    = tid % K1_;
  int s2   = (tid / K1_) % HW_;
  int item = tid / ((size_t)K1_ * HW_);
  int Y = s2 / W_, X = s2 % W_;
  int c = k / 9, t = k % 9, u = t / 3, v = t % 3;
  int yy = Y + u - 1, xx = X + v - 1;
  float val = 0.f;
  if ((unsigned)yy < (unsigned)H_ && (unsigned)xx < (unsigned)W_)
    val = src[(((size_t)item * C_ + c) * H_ + yy) * W_ + xx];
  R[tid] = (_Float16)val;
}

// ---------------- GEMM 1: scores, double-buffered, writes transposed Yt[s][p] ----------------
__global__ void __launch_bounds__(256) k_gemm1(const _Float16* __restrict__ Wn,
                                               const _Float16* __restrict__ Xc,
                                               float* __restrict__ Yt) {
  __shared__ _Float16 As[2][128 * LDK];
  __shared__ _Float16 Bs[2][128 * LDK];
  int item = blockIdx.z;
  int m0 = blockIdx.y * 128;
  int n0 = blockIdx.x * 128;
  const _Float16* A  = Wn + (size_t)item * L_ * K1_;
  const _Float16* Bm = Xc + (size_t)item * L_ * K1_;
  int t = threadIdx.x;
  int lane = t & 31, wv = t >> 5;
  int wm = wv >> 2, wn = wv & 3;
  int lh = lane & 15, hi = lane >> 4;
  v8f acc[4][2] = {};
  int lrow = t >> 2;
  int lkp  = (t & 3) * 8;

  auto load_tiles = [&](int kc, int bi) {
    int k0 = kc * 32;
#pragma unroll
    for (int pass = 0; pass < 2; ++pass) {
      int rr = lrow + pass * 64;
      cp16(&As[bi][rr * LDK + lkp], &A[(size_t)(m0 + rr) * K1_ + k0 + lkp]);
      cp16(&Bs[bi][rr * LDK + lkp], &Bm[(size_t)(n0 + rr) * K1_ + k0 + lkp]);
    }
  };

  load_tiles(0, 0);
  for (int kc = 0; kc < 18; ++kc) {
    int cur = kc & 1;
    cp_commit();
    __syncthreads();
    if (kc < 17) load_tiles(kc + 1, cur ^ 1);
    v16h bf[2];
#pragma unroll
    for (int ni = 0; ni < 2; ++ni) {
      int n  = wn * 32 + ni * 16 + lh;
      int kb = hi * 8;
      bf[ni] = cat8(*(v8h*)&Bs[cur][n * LDK + kb], *(v8h*)&Bs[cur][n * LDK + kb + 16]);
    }
#pragma unroll
    for (int mi = 0; mi < 4; ++mi) {
      int m  = wm * 64 + mi * 16 + lh;
      int ka = hi * 8;
      v16h af = cat8(*(v8h*)&As[cur][m * LDK + ka], *(v8h*)&As[cur][m * LDK + ka + 16]);
#pragma unroll
      for (int ni = 0; ni < 2; ++ni)
        acc[mi][ni] = wmma_f16(af, bf[ni], acc[mi][ni]);
    }
  }
  float* Yti = Yt + (size_t)item * L_ * L_;
#pragma unroll
  for (int mi = 0; mi < 4; ++mi) {
#pragma unroll
    for (int ni = 0; ni < 2; ++ni) {
      int ng = n0 + wn * 32 + ni * 16 + lh;
      int mg = m0 + wm * 64 + mi * 16 + 8 * hi;
      float4 lo  = make_float4(acc[mi][ni][0], acc[mi][ni][1], acc[mi][ni][2], acc[mi][ni][3]);
      float4 hi4 = make_float4(acc[mi][ni][4], acc[mi][ni][5], acc[mi][ni][6], acc[mi][ni][7]);
      *(float4*)&Yti[(size_t)ng * L_ + mg]     = lo;
      *(float4*)&Yti[(size_t)ng * L_ + mg + 4] = hi4;
    }
  }
}

// ---------------- fuse pass 1 (diagonal blur, transposed space) ----------------
__global__ void k_fuse1(const float* __restrict__ Yt, float* __restrict__ Z1t) {
  size_t tid = (size_t)blockIdx.x * blockDim.x + threadIdx.x;
  if (tid >= (size_t)BN * L_ * L_) return;
  int a  = tid % L_;
  int bb = (tid / L_) % L_;
  size_t base = (tid / ((size_t)L_ * L_)) * (size_t)L_ * L_;
  const float* Y = Yt + base;
  float v = Y[(size_t)bb * L_ + a];
  if (bb > 0 && a > 0)             v += Y[(size_t)(bb - 1) * L_ + (a - 1)];
  if (bb < L_ - 1 && a < L_ - 1)   v += Y[(size_t)(bb + 1) * L_ + (a + 1)];
  Z1t[tid] = v;
}

// ---------------- fuse pass 2 (index-swapped) + softmax over p ----------------
__global__ void __launch_bounds__(256) k_fuse2sm(const float* __restrict__ Z1t,
                                                 _Float16* __restrict__ Asm) {
  __shared__ float rows[3][L_];
  __shared__ float red[8];
  __shared__ float bc;
  int s    = blockIdx.x % L_;
  int item = blockIdx.x / L_;
  const float* Z = Z1t + (size_t)item * L_ * L_;
  int t = threadIdx.x;
  bool vs[3]; int cb[3];
#pragma unroll
  for (int d = 0; d < 3; ++d) {
    int q = swp(s) + d - 1;
    vs[d] = (q >= 0 && q < L_);
    cb[d] = vs[d] ? swp(q) : 0;
  }
  for (int d = 0; d < 3; ++d) {
    if (vs[d]) {
      const float* src = Z + (size_t)cb[d] * L_;
      for (int idx = t; idx < L_; idx += 256) rows[d][idx] = src[idx];
    }
  }
  __syncthreads();
  float tv[9];
  float mx = -3.4e38f;
#pragma unroll
  for (int j = 0; j < 9; ++j) {
    int p  = t + 256 * j;
    int sp = swp(p);
    float T = 0.f;
#pragma unroll
    for (int d = 0; d < 3; ++d) {
      if (!vs[d]) continue;
      int q = sp + d - 1;
      if (q < 0 || q >= L_) continue;
      T += rows[d][swp(q)];
    }
    tv[j] = T * 10.0f;
    mx = fmaxf(mx, tv[j]);
  }
#pragma unroll
  for (int off = 16; off > 0; off >>= 1) mx = fmaxf(mx, __shfl_xor(mx, off, 32));
  if ((t & 31) == 0) red[t >> 5] = mx;
  __syncthreads();
  if (t == 0) {
    float m = red[0];
    for (int i = 1; i < 8; ++i) m = fmaxf(m, red[i]);
    bc = m;
  }
  __syncthreads();
  float M = bc;
  float se = 0.f;
#pragma unroll
  for (int j = 0; j < 9; ++j) { tv[j] = __expf(tv[j] - M); se += tv[j]; }
#pragma unroll
  for (int off = 16; off > 0; off >>= 1) se += __shfl_xor(se, off, 32);
  __syncthreads();
  if ((t & 31) == 0) red[t >> 5] = se;
  __syncthreads();
  if (t == 0) {
    float ssum = 0.f;
    for (int i = 0; i < 8; ++i) ssum += red[i];
    bc = ssum;
  }
  __syncthreads();
  float inv = 1.f / bc;
  _Float16* dst = Asm + ((size_t)item * L_ + s) * L_;
#pragma unroll
  for (int j = 0; j < 9; ++j) dst[t + 256 * j] = (_Float16)(tv[j] * inv);
}

// ---------------- deconv as one 288-tile GEMM stream (4 taps x 72 K-chunks) ----------------
__global__ void __launch_bounds__(256) k_gemm_deconv(const _Float16* __restrict__ RWt,
                                                     const _Float16* __restrict__ Asm,
                                                     float* __restrict__ Yd) {
  __shared__ _Float16 As[2][64 * LDK];
  __shared__ _Float16 Bs[2][256 * LDK];
  int item  = blockIdx.z;
  int phase = blockIdx.y;
  int ey = phase >> 1, ex = phase & 1;
  int n0 = blockIdx.x * 256;
  int t = threadIdx.x;
  int lane = t & 31, wv = t >> 5;
  int lh = lane & 15, hi = lane >> 4;
  int nw = wv * 32;
  const _Float16* Am = Asm + (size_t)item * L_ * L_;
  const v8h hzero = {};
  v8f acc[4][2] = {};
  int jn = n0 + t;
  int iy = jn / 48, ix = jn % 48;
  int lrow = t >> 2;
  int lkp  = (t & 3) * 8;

  auto load_tiles = [&](int ti, int bi) {
    int ab = ti / 72, kc = ti % 72;
    int a = ab >> 1, bb2 = ab & 1;
    int u = ey + 2 * a, v = ex + 2 * bb2;
    int tap = u * 4 + v;
    const _Float16* Ag = RWt + (((size_t)item * 16 + tap) * C_) * L_;
    int k0 = kc * 32;
    cp16(&As[bi][lrow * LDK + lkp], &Ag[(size_t)lrow * L_ + k0 + lkp]);
    int h = iy + ey - 1 + a;
    int w = ix + ex - 1 + bb2;
    if (h >= 0 && h < 48 && w >= 0 && w < 48) {
      size_t srow = (size_t)(h * 48 + w) * L_;
#pragma unroll
      for (int q = 0; q < 4; ++q)
        cp16(&Bs[bi][t * LDK + q * 8], &Am[srow + k0 + q * 8]);
    } else {
#pragma unroll
      for (int q = 0; q < 4; ++q)
        *(v8h*)&Bs[bi][t * LDK + q * 8] = hzero;
    }
  };

  load_tiles(0, 0);
  for (int ti = 0; ti < 288; ++ti) {
    int cur = ti & 1;
    cp_commit();
    __syncthreads();
    if (ti < 287) load_tiles(ti + 1, cur ^ 1);
    v16h bf[2];
#pragma unroll
    for (int ni = 0; ni < 2; ++ni) {
      int n  = nw + ni * 16 + lh;
      int kb = hi * 8;
      bf[ni] = cat8(*(v8h*)&Bs[cur][n * LDK + kb], *(v8h*)&Bs[cur][n * LDK + kb + 16]);
    }
#pragma unroll
    for (int mi = 0; mi < 4; ++mi) {
      int m  = mi * 16 + lh;
      int ka = hi * 8;
      v16h af = cat8(*(v8h*)&As[cur][m * LDK + ka], *(v8h*)&As[cur][m * LDK + ka + 16]);
#pragma unroll
      for (int ni = 0; ni < 2; ++ni)
        acc[mi][ni] = wmma_f16(af, bf[ni], acc[mi][ni]);
    }
  }
  float* Yo = Yd + (size_t)item * C_ * HW_;
#pragma unroll
  for (int mi = 0; mi < 4; ++mi) {
#pragma unroll
    for (int ni = 0; ni < 2; ++ni) {
      int ng = n0 + nw + ni * 16 + lh;
      int oy = 2 * (ng / 48) + ey, ox = 2 * (ng % 48) + ex;
      int cb2 = mi * 16 + 8 * hi;
#pragma unroll
      for (int r = 0; r < 8; ++r)
        Yo[(size_t)(cb2 + r) * HW_ + oy * W_ + ox] = acc[mi][ni][r];
    }
  }
}

// ---------------- 3x3 conv as GEMM (M=64, K=576, N=9216) + bias ----------------
__global__ void __launch_bounds__(256) k_gemm_conv(const _Float16* __restrict__ Wh,
                                                   const _Float16* __restrict__ R,
                                                   const float* __restrict__ bias,
                                                   float* __restrict__ dst) {
  __shared__ _Float16 As[2][64 * LDK];
  __shared__ _Float16 Bs[2][256 * LDK];
  int item = blockIdx.y;
  int n0 = blockIdx.x * 256;
  int t = threadIdx.x;
  int lane = t & 31, wv = t >> 5;
  int lh = lane & 15, hi = lane >> 4;
  int nw = wv * 32;
  v8f acc[4][2] = {};
  int lrow = t >> 2, lkp = (t & 3) * 8;
  const _Float16* Rr = R + ((size_t)item * HW_ + n0) * K1_;

  auto load_tiles = [&](int kc, int bi) {
    int k0 = kc * 32;
    cp16(&As[bi][lrow * LDK + lkp], &Wh[(size_t)lrow * K1_ + k0 + lkp]);
#pragma unroll
    for (int q = 0; q < 4; ++q)
      cp16(&Bs[bi][t * LDK + q * 8], &Rr[(size_t)t * K1_ + k0 + q * 8]);
  };

  load_tiles(0, 0);
  for (int kc = 0; kc < 18; ++kc) {
    int cur = kc & 1;
    cp_commit();
    __syncthreads();
    if (kc < 17) load_tiles(kc + 1, cur ^ 1);
    v16h bf[2];
#pragma unroll
    for (int ni = 0; ni < 2; ++ni) {
      int n  = nw + ni * 16 + lh;
      int kb = hi * 8;
      bf[ni] = cat8(*(v8h*)&Bs[cur][n * LDK + kb], *(v8h*)&Bs[cur][n * LDK + kb + 16]);
    }
#pragma unroll
    for (int mi = 0; mi < 4; ++mi) {
      int m  = mi * 16 + lh;
      int ka = hi * 8;
      v16h af = cat8(*(v8h*)&As[cur][m * LDK + ka], *(v8h*)&As[cur][m * LDK + ka + 16]);
#pragma unroll
      for (int ni = 0; ni < 2; ++ni)
        acc[mi][ni] = wmma_f16(af, bf[ni], acc[mi][ni]);
    }
  }
#pragma unroll
  for (int mi = 0; mi < 4; ++mi) {
#pragma unroll
    for (int ni = 0; ni < 2; ++ni) {
      int ng  = n0 + nw + ni * 16 + lh;
      int cb2 = mi * 16 + 8 * hi;
#pragma unroll
      for (int r = 0; r < 8; ++r)
        dst[((size_t)item * C_ + cb2 + r) * HW_ + ng] = acc[mi][ni][r] + bias[cb2 + r];
    }
  }
}

// ---------------- host launcher ----------------
extern "C" void kernel_launch(void* const* d_in, const int* in_sizes, int n_in,
                              void* d_out, int out_size, void* d_ws, size_t ws_size,
                              hipStream_t stream) {
  (void)in_sizes; (void)n_in; (void)out_size; (void)ws_size;
  const float* f  = (const float*)d_in[0];
  const float* b  = (const float*)d_in[1];
  const float* w1 = (const float*)d_in[2];
  const float* b1 = (const float*)d_in[3];
  const float* w2 = (const float*)d_in[4];
  const float* b2 = (const float*)d_in[5];

  char* ws = (char*)d_ws;
  size_t off = 0;
  auto alloc = [&](size_t bytes) -> void* {
    void* p = ws + off;
    off += (bytes + 255) & ~(size_t)255;
    return p;
  };
  _Float16* Xc  = (_Float16*)alloc((size_t)BN * L_ * K1_ * 2);
  _Float16* Wn  = (_Float16*)alloc((size_t)BN * L_ * K1_ * 2);
  _Float16* RWt = (_Float16*)alloc((size_t)BN * 16 * C_ * L_ * 2);
  _Float16* Wh  = (_Float16*)alloc((size_t)2 * C_ * K1_ * 2);
  float*    Yt  = (float*)alloc((size_t)BN * L_ * L_ * 4);
  float*    Z1t = (float*)alloc((size_t)BN * L_ * L_ * 4);
  _Float16* Amx = (_Float16*)alloc((size_t)BN * L_ * L_ * 2);
  float*    Yd  = (float*)alloc((size_t)BN * C_ * HW_ * 4);
  float*    H1  = (float*)alloc((size_t)BN * C_ * HW_ * 4);
  _Float16* Rm  = (_Float16*)alloc((size_t)BN * HW_ * K1_ * 2);

  k_build_xcol<<<20736, 256, 0, stream>>>(f, Xc);
  k_build_wn<<<1152, 256, 0, stream>>>(b, Wn);
  k_build_rwt<<<36864, 256, 0, stream>>>(b, RWt);
  k_wcvt<<<288, 256, 0, stream>>>(w1, w2, Wh);

  k_gemm1<<<dim3(18, 18, BN), 256, 0, stream>>>(Wn, Xc, Yt);
  k_fuse1<<<82944, 256, 0, stream>>>(Yt, Z1t);
  k_fuse2sm<<<BN * L_, 256, 0, stream>>>(Z1t, Amx);
  k_gemm_deconv<<<dim3(9, 4, BN), 256, 0, stream>>>(RWt, Amx, Yd);

  k_im2row<<<82944, 256, 0, stream>>>(Yd, Rm);
  k_gemm_conv<<<dim3(36, BN), 256, 0, stream>>>(Wh, Rm, b1, H1);
  k_im2row<<<82944, 256, 0, stream>>>(H1, Rm);
  k_gemm_conv<<<dim3(36, BN), 256, 0, stream>>>(Wh + (size_t)C_ * K1_, Rm, b2, (float*)d_out);
}
